// TimeSeriesModel2_64407329571297
// MI455X (gfx1250) — matrix operand order repaired and checked by reading, not compiled, not run on hardware
//
#include <hip/hip_runtime.h>
#include <hip/hip_bf16.h>
#include <stdint.h>

// ---------------------------------------------------------------------------
// mLSTM: heavy part = X[1024,37888] @ [Wmx|Wgx][37888,500]  (~38.8 GFLOP)
// v_wmma_f32_16x16x32_bf16, split-K for occupancy/bandwidth, deterministic
// fixed-order reduction, persistent-block recurrence.
// ---------------------------------------------------------------------------

typedef __attribute__((ext_vector_type(16))) __bf16 v16bf;
typedef __attribute__((ext_vector_type(8)))  float  v8f;

#define IN_DIM 37888
#define H_DIM  100
#define G_DIM  400
#define NCOL   500      // H + 4H real columns
#define NPAD   512      // padded N
#define MTOT   1024     // B*S
#define TM     128
#define TN     128
#define KT     32
#define LDA    40       // padded LDS stride (halves)
#define KSTEPS_TOTAL (IN_DIM / KT)   // 1184

union FragU { uint4 u[2]; v16bf v; };

// fp32 -> bf16 via native converts (lets clang pick v_cvt_pk_bf16_f32)
__device__ __forceinline__ FragU cvt16(const float* f) {
  FragU r;
#pragma unroll
  for (int i = 0; i < 16; ++i) r.v[i] = (__bf16)f[i];
  return r;
}

__global__ __launch_bounds__(256)
void proj_gemm_wmma(const float* __restrict__ X,
                    const float* __restrict__ Wmx,
                    const float* __restrict__ Wgx,
                    float* __restrict__ partial,   // [ksplit][MTOT][NPAD]
                    int ksteps) {
  __shared__ unsigned short lds_a[2][TM * LDA];   // A tile, row-major bf16
  __shared__ unsigned short lds_b[2][TN * LDA];   // B tile, column-major bf16

  const int tid   = threadIdx.x;
  const int lane  = tid & 31;
  const int wave  = tid >> 5;
  const int waveM = wave >> 2;        // 0..1 -> 64 rows each
  const int waveN = wave & 3;         // 0..3 -> 32 cols each
  const int mBase = blockIdx.y * TM;
  const int nBase = blockIdx.x * TN;
  const int kBase = blockIdx.z * ksteps * KT;
  float* __restrict__ cdst = partial + (size_t)blockIdx.z * MTOT * NPAD;

  const v8f vzero = {0.f, 0.f, 0.f, 0.f, 0.f, 0.f, 0.f, 0.f};
  v8f acc[4][2];
#pragma unroll
  for (int mi = 0; mi < 4; ++mi)
#pragma unroll
    for (int ni = 0; ni < 2; ++ni) acc[mi][ni] = vzero;

  const int r   = tid >> 1;   // row (A) / column (B) within tile: 0..127
  const int seg = tid & 1;    // which 16-wide K segment

  auto stage = [&](int buf, int k0) {
    // ---- A tile: 128 rows x 32 K (fp32 -> bf16) ----
    {
      const float* src = X + (size_t)(mBase + r) * IN_DIM + k0 + seg * 16;
      __builtin_prefetch(src + 2 * KT, 0, 1);     // global_prefetch_b8, 2 tiles ahead
      float tmp[16];
      float4 f0 = ((const float4*)src)[0];
      float4 f1 = ((const float4*)src)[1];
      float4 f2 = ((const float4*)src)[2];
      float4 f3 = ((const float4*)src)[3];
      tmp[0]=f0.x; tmp[1]=f0.y; tmp[2]=f0.z; tmp[3]=f0.w;
      tmp[4]=f1.x; tmp[5]=f1.y; tmp[6]=f1.z; tmp[7]=f1.w;
      tmp[8]=f2.x; tmp[9]=f2.y; tmp[10]=f2.z; tmp[11]=f2.w;
      tmp[12]=f3.x; tmp[13]=f3.y; tmp[14]=f3.z; tmp[15]=f3.w;
      FragU fr = cvt16(tmp);
      unsigned short* dst = &lds_a[buf][r * LDA + seg * 16];
      ((uint4*)dst)[0] = fr.u[0];
      ((uint4*)dst)[1] = fr.u[1];
    }
    // ---- B tile: 32 K x 128 cols, stored column-major in LDS ----
    // lanes with consecutive r hit consecutive global addresses -> coalesced
    {
      const int c   = nBase + r;          // combined column id
      const int kk0 = k0 + seg * 16;
      float tmp[16];
      if (c < H_DIM) {
#pragma unroll
        for (int i = 0; i < 16; ++i)
          tmp[i] = Wmx[(size_t)(kk0 + i) * H_DIM + c];
      } else if (c < NCOL) {
#pragma unroll
        for (int i = 0; i < 16; ++i)
          tmp[i] = Wgx[(size_t)(kk0 + i) * G_DIM + (c - H_DIM)];
      } else {
#pragma unroll
        for (int i = 0; i < 16; ++i) tmp[i] = 0.f;
      }
      FragU fr = cvt16(tmp);
      unsigned short* dst = &lds_b[buf][r * LDA + seg * 16];
      ((uint4*)dst)[0] = fr.u[0];
      ((uint4*)dst)[1] = fr.u[1];
    }
  };

  auto compute = [&](int buf) {
    FragU bfr[2];
#pragma unroll
    for (int ni = 0; ni < 2; ++ni) {
      const int col = waveN * 32 + ni * 16 + (lane & 15);
      const unsigned short* p = &lds_b[buf][col * LDA + (lane >> 4) * 16];
      bfr[ni].u[0] = *(const uint4*)(p);
      bfr[ni].u[1] = *(const uint4*)(p + 8);
    }
#pragma unroll
    for (int mi = 0; mi < 4; ++mi) {
      const int row = waveM * 64 + mi * 16 + (lane & 15);
      const int kb  = (lane >> 4) * 16;   // lanes<16: K{0-15}; >=16: K{16-31}
      const unsigned short* p = &lds_a[buf][row * LDA + kb];
      FragU afr;
      afr.u[0] = *(const uint4*)(p);
      afr.u[1] = *(const uint4*)(p + 8);
#pragma unroll
      for (int ni = 0; ni < 2; ++ni) {
        acc[mi][ni] = __builtin_amdgcn_wmma_f32_16x16x32_bf16(
            false, afr.v, false, bfr[ni].v, (short)0, acc[mi][ni], false, false);
      }
    }
  };

  stage(0, kBase);
  __syncthreads();
  for (int kt = 0; kt < ksteps; ++kt) {
    const int cur = kt & 1;
    if (kt + 1 < ksteps) stage(cur ^ 1, kBase + (kt + 1) * KT);
    compute(cur);
    __syncthreads();
  }

  // Epilogue: write padded partial tile (no branches; cols 500..511 are junk pad)
#pragma unroll
  for (int mi = 0; mi < 4; ++mi) {
    const int row0 = mBase + waveM * 64 + mi * 16 + (lane >> 4) * 8;
#pragma unroll
    for (int ni = 0; ni < 2; ++ni) {
      const int col = nBase + waveN * 32 + ni * 16 + (lane & 15);
#pragma unroll
      for (int e = 0; e < 8; ++e) {
        cdst[(size_t)(row0 + e) * NPAD + col] = acc[mi][ni][e];
      }
    }
  }
}

// ---------------------------------------------------------------------------
// Fixed-order split-K reduction + bias fold: deterministic accumulation.
// ---------------------------------------------------------------------------
__global__ __launch_bounds__(256)
void reduce_bias(const float* __restrict__ partial, const float* __restrict__ bg,
                 float* __restrict__ XM, float* __restrict__ XG, int ksplit) {
  const int idx = blockIdx.x * blockDim.x + threadIdx.x;
  if (idx >= MTOT * NCOL) return;
  const int row = idx / NCOL, col = idx % NCOL;
  float a = 0.f;
  for (int k = 0; k < ksplit; ++k)
    a += partial[(size_t)k * MTOT * NPAD + (size_t)row * NPAD + col];
  if (col < H_DIM)
    XM[row * H_DIM + col] = a;
  else
    XG[row * G_DIM + (col - H_DIM)] = a + bg[col - H_DIM];
}

// ---------------------------------------------------------------------------
// Sequential recurrence + regressor head. One block, state in LDS.
// ---------------------------------------------------------------------------
__device__ __forceinline__ float sigf(float x) { return 1.f / (1.f + expf(-x)); }

__global__ __launch_bounds__(512)
void mlstm_recurrence(const float* __restrict__ XM, const float* __restrict__ XG,
                      const float* __restrict__ Wmh, const float* __restrict__ Wgm,
                      const float* __restrict__ W1, const float* __restrict__ b1,
                      const float* __restrict__ W2, const float* __restrict__ b2,
                      float* __restrict__ out) {
  __shared__ float sh[32 * H_DIM];
  __shared__ float sc[32 * H_DIM];
  __shared__ float sm[32 * H_DIM];
  __shared__ float sz[32 * 32];
  const int tid = threadIdx.x;

  for (int i = tid; i < 32 * H_DIM; i += 512) { sh[i] = 0.f; sc[i] = 0.f; }
  __syncthreads();

  for (int t = 0; t < 32; ++t) {
    // m = xm_t * (h @ W_mh)
    for (int idx = tid; idx < 32 * H_DIM; idx += 512) {
      const int b = idx / H_DIM, j = idx % H_DIM;
      float a = 0.f;
#pragma unroll 4
      for (int k = 0; k < H_DIM; ++k) a += sh[b * H_DIM + k] * Wmh[k * H_DIM + j];
      sm[idx] = XM[(size_t)(b * 32 + t) * H_DIM + j] * a;
    }
    __syncthreads();
    // gates = xg_t + m @ W_gm ; cell update
    for (int idx = tid; idx < 32 * H_DIM; idx += 512) {
      const int b = idx / H_DIM, j = idx % H_DIM;
      const float* xg = XG + (size_t)(b * 32 + t) * G_DIM;   // bias already folded in
      float gi = xg[j], gf = xg[H_DIM + j], go = xg[2 * H_DIM + j], gg = xg[3 * H_DIM + j];
#pragma unroll 2
      for (int k = 0; k < H_DIM; ++k) {
        const float mv = sm[b * H_DIM + k];
        const float* w = Wgm + (size_t)k * G_DIM;
        gi += mv * w[j];
        gf += mv * w[H_DIM + j];
        go += mv * w[2 * H_DIM + j];
        gg += mv * w[3 * H_DIM + j];
      }
      const float cc = sigf(gf) * sc[idx] + sigf(gi) * tanhf(gg);
      const float hh = sigf(go) * tanhf(cc);
      sc[idx] = cc;
      sh[idx] = hh;
    }
    __syncthreads();
  }

  // outputs: h_n [3200], c_n [3200]
  for (int i = tid; i < 32 * H_DIM; i += 512) {
    out[i] = sh[i];
    out[3200 + i] = sc[i];
  }
  // z = relu(h @ W1 + b1)
  for (int idx = tid; idx < 32 * 32; idx += 512) {
    const int b = idx >> 5, j = idx & 31;
    float a = b1[j];
#pragma unroll 4
    for (int k = 0; k < H_DIM; ++k) a += sh[b * H_DIM + k] * W1[k * 32 + j];
    sz[idx] = a > 0.f ? a : 0.f;
  }
  __syncthreads();
  // pred = sigmoid(z @ W2 + b2)
  for (int idx = tid; idx < 64; idx += 512) {
    const int b = idx >> 1, j = idx & 1;
    float a = b2[j];
#pragma unroll
    for (int k = 0; k < 32; ++k) a += sz[b * 32 + k] * W2[k * 2 + j];
    out[6400 + idx] = sigf(a);
  }
}

extern "C" void kernel_launch(void* const* d_in, const int* in_sizes, int n_in,
                              void* d_out, int out_size, void* d_ws, size_t ws_size,
                              hipStream_t stream) {
  (void)in_sizes; (void)n_in; (void)out_size;
  const float* X   = (const float*)d_in[0];
  const float* Wmx = (const float*)d_in[1];
  const float* Wmh = (const float*)d_in[2];
  const float* Wgx = (const float*)d_in[3];
  const float* Wgm = (const float*)d_in[4];
  const float* bg  = (const float*)d_in[5];
  const float* W1  = (const float*)d_in[6];
  const float* b1  = (const float*)d_in[7];
  const float* W2  = (const float*)d_in[8];
  const float* b2  = (const float*)d_in[9];
  float* out = (float*)d_out;

  const size_t partElems = (size_t)MTOT * NPAD;           // one padded C copy
  const size_t fixedElems = (size_t)MTOT * (H_DIM + G_DIM);
  int ksplit = 8;                                         // 256 workgroups
  if (ws_size < (8 * partElems + fixedElems) * sizeof(float)) ksplit = 1;

  float* partial = (float*)d_ws;                          // [ksplit][1024][512]
  float* XM = partial + (size_t)ksplit * partElems;       // [1024,100]
  float* XG = XM + (size_t)MTOT * H_DIM;                  // [1024,400] (bias folded)

  const int ksteps = KSTEPS_TOTAL / ksplit;
  dim3 gridA(NPAD / TN, MTOT / TM, ksplit);               // 4 x 8 x ksplit
  proj_gemm_wmma<<<gridA, dim3(256), 0, stream>>>(X, Wmx, Wgx, partial, ksteps);

  const int redThreads = 256;
  const int redBlocks = (MTOT * NCOL + redThreads - 1) / redThreads;
  reduce_bias<<<dim3(redBlocks), dim3(redThreads), 0, stream>>>(partial, bg, XM, XG, ksplit);

  mlstm_recurrence<<<dim3(1), dim3(512), 0, stream>>>(XM, XG, Wmh, Wgm,
                                                      W1, b1, W2, b2, out);
}
